// Self_Attention_4750233830173
// MI455X (gfx1250) — matrix operand order repaired and checked by reading, not compiled, not run on hardware
//
#include <hip/hip_runtime.h>

// ---------------------------------------------------------------------------
// Self-attention (softmax over key axis) for B=4, C=256, N=4096, D=32
// Pass W: convert fused weight matrix [320][256] to bf16
// Pass 0: WMMA projections -> bf16 qT [b][n][d], kT [b][n][d], v [b][c][n]
//         (x staged to LDS pre-packed in WMMA B-operand image layout)
// Pass 1: WMMA bf16 Q^T K tiles + online softmax stats m[j], l[j]
// Pass 2: WMMA flash-style  O = V * exp(S - m) / l,  out = x + gamma * O
//         (P tile exchanged through LDS as packed b128 B-operand image)
// ---------------------------------------------------------------------------

typedef __bf16 bf16_t;
typedef bf16_t v16bf __attribute__((ext_vector_type(16)));
typedef float  v8f   __attribute__((ext_vector_type(8)));

struct alignas(16) B128 { unsigned u[4]; };

union V16 {
  v16bf v;
  B128  q[2];
  unsigned u[8];
};

#define NB   4
#define CC   256
#define NN   4096
#define DD   32
#define WROWS 320
#define L2E  1.4426950408889634f

__device__ __forceinline__ float fast_exp2(float x) {
#if __has_builtin(__builtin_amdgcn_exp2f)
  return __builtin_amdgcn_exp2f(x);
#else
  return exp2f(x);
#endif
}

__device__ __forceinline__ unsigned short f32_to_bf16(float f) {
  union { float f; unsigned u; } c; c.f = f;
  unsigned u = c.u;
  u += 0x7FFFu + ((u >> 16) & 1u);   // round-to-nearest-even
  return (unsigned short)(u >> 16);
}

__device__ __forceinline__ unsigned pack_bf16(float a, float b) {
  return (unsigned)f32_to_bf16(a) | ((unsigned)f32_to_bf16(b) << 16);
}

__device__ __forceinline__ B128 ld128(const unsigned short* p) {
  return *(const B128*)p;
}

// ---------------------------------------------------------------------------
// Pass W: fuse wq/wk/wv into one bf16 matrix [320][256] row-major.
// ---------------------------------------------------------------------------
__global__ __launch_bounds__(256) void wcvt_kernel(
    const float* __restrict__ wq, const float* __restrict__ wk,
    const float* __restrict__ wv, unsigned short* __restrict__ wbf) {
  int idx = blockIdx.x * 256 + threadIdx.x;   // 0 .. 320*256-1
  int row = idx >> 8;
  int cin = idx & 255;
  float v;
  if (row < 32)      v = wq[row * CC + cin];
  else if (row < 64) v = wk[(row - 32) * CC + cin];
  else               v = wv[(row - 64) * CC + cin];
  wbf[idx] = f32_to_bf16(v);
}

// ---------------------------------------------------------------------------
// Pass 0: WMMA projections.  grid (N/128, B), block 512 (16 waves).
// Wave w: n-tile nt = w&7, output-half oh = w>>3 (10 oc-tiles of 16 each).
// x chunk [32 cin x 128 n] staged to LDS as packed B-operand image:
//   BX[n][p] = pack(x[c0+2p][n], x[c0+2p+1][n])   (row stride 20 dwords)
// ---------------------------------------------------------------------------
__global__ __launch_bounds__(512) void proj_kernel(
    const float* __restrict__ x,
    const float* __restrict__ bq, const float* __restrict__ bk,
    const float* __restrict__ bvp,
    const unsigned short* __restrict__ wbf,
    unsigned short* __restrict__ qT, unsigned short* __restrict__ kT,
    unsigned short* __restrict__ vB) {
  __shared__ __align__(16) unsigned BX[128 * 20];

  int b  = blockIdx.y;
  int n0 = blockIdx.x * 128;
  int t  = threadIdx.x;
  int w = t >> 5, lane = t & 31;
  int col = lane & 15, hi = lane >> 4;
  int nt = w & 7, oh = w >> 3;
  const float* xb = x + (size_t)b * CC * NN;

  v8f acc[10];
#pragma unroll
  for (int i = 0; i < 10; ++i)
    acc[i] = (v8f){0.f, 0.f, 0.f, 0.f, 0.f, 0.f, 0.f, 0.f};

  int sn  = t & 127;   // staging n-column
  int sp0 = t >> 7;    // staging pair base 0..3

  for (int c0 = 0; c0 < CC; c0 += 32) {
    // ---- stage x chunk into LDS (packed bf16 pairs, B-image layout) ----
#pragma unroll
    for (int i = 0; i < 4; ++i) {
      int p = sp0 + i * 4;                       // cin pair index 0..15
      float a0 = xb[(size_t)(c0 + 2 * p) * NN + n0 + sn];
      float a1 = xb[(size_t)(c0 + 2 * p + 1) * NN + n0 + sn];
      BX[sn * 20 + p] = pack_bf16(a0, a1);
    }
    __syncthreads();

    // ---- B operand for this wave's n-tile: 2x ds_load_b128 ----
    V16 Bx;
    {
      const B128* pp = (const B128*)&BX[(nt * 16 + col) * 20 + hi * 8];
      Bx.q[0] = pp[0];
      Bx.q[1] = pp[1];
    }
    // ---- 10 output tiles, A = weight rows (A-layout direct from global) ----
#pragma unroll
    for (int ot = 0; ot < 10; ++ot) {
      int oc0 = (oh * 10 + ot) * 16;
      V16 Aw;
      const unsigned short* wr = wbf + (size_t)(oc0 + col) * CC + c0 + hi * 16;
      Aw.q[0] = ld128(wr);
      Aw.q[1] = ld128(wr + 8);
      acc[ot] = __builtin_amdgcn_wmma_f32_16x16x32_bf16(false, Aw.v, false, Bx.v,
                                                        (short)0, acc[ot],
                                                        false, false);
    }
    __syncthreads();
  }

  // ---- epilogue: add bias, convert, scatter to qT / kT / vB ----
  int n = n0 + nt * 16 + col;
#pragma unroll
  for (int ot = 0; ot < 10; ++ot) {
    int oc_t = (oh * 10 + ot) * 16;   // tile lies entirely in q, k, or v
#pragma unroll
    for (int r = 0; r < 8; ++r) {
      int oc = oc_t + hi * 8 + r;
      float bias = (oc < 32) ? bq[oc] : (oc < 64 ? bk[oc - 32] : bvp[oc - 64]);
      unsigned short hv = f32_to_bf16(acc[ot][r] + bias);
      if (oc < 32)      qT[((size_t)b * NN + n) * DD + oc] = hv;
      else if (oc < 64) kT[((size_t)b * NN + n) * DD + (oc - 32)] = hv;
      else              vB[((size_t)b * CC + (oc - 64)) * NN + n] = hv;
    }
  }
}

// ---------------------------------------------------------------------------
// Pass 1: softmax stats over the n axis (axis 1 of S[b,i,j]).
// grid (N/128, B), block 256 (8 waves); wave w owns columns j0 = blk*128+w*16.
// ---------------------------------------------------------------------------
__global__ __launch_bounds__(256) void stats_kernel(
    const unsigned short* __restrict__ qT, const unsigned short* __restrict__ kT,
    float* __restrict__ mOut, float* __restrict__ lOut) {
  int b    = blockIdx.y;
  int w    = threadIdx.x >> 5;
  int lane = threadIdx.x & 31;
  int col  = lane & 15;
  int hi   = lane >> 4;
  int j0   = blockIdx.x * 128 + w * 16;

  // B operand: k tile [d=32 x j=16], resident for the whole n loop.
  V16 Bk;
  {
    const unsigned short* kr = kT + ((size_t)b * NN + j0 + col) * DD + hi * 16;
    Bk.q[0] = ld128(kr);
    Bk.q[1] = ld128(kr + 8);
  }

  float m_run = -1e30f, l_run = 0.0f;
  for (int n0 = 0; n0 < NN; n0 += 16) {
    V16 Aq;
    {
      const unsigned short* qr = qT + ((size_t)b * NN + n0 + col) * DD + hi * 16;
      Aq.q[0] = ld128(qr);
      Aq.q[1] = ld128(qr + 8);
    }
    v8f S = {0.f, 0.f, 0.f, 0.f, 0.f, 0.f, 0.f, 0.f};
    S = __builtin_amdgcn_wmma_f32_16x16x32_bf16(false, Aq.v, false, Bk.v,
                                                (short)0, S, false, false);
    float mc = S[0];
#pragma unroll
    for (int r = 1; r < 8; ++r) mc = fmaxf(mc, S[r]);
    float m_new = fmaxf(m_run, mc);
    float sum = 0.f;
#pragma unroll
    for (int r = 0; r < 8; ++r) sum += fast_exp2((S[r] - m_new) * L2E);
    l_run = l_run * fast_exp2((m_run - m_new) * L2E) + sum;
    m_run = m_new;
  }
  // combine the two lane halves (same j, disjoint n rows)
  float m_o = __shfl_xor(m_run, 16, 32);
  float l_o = __shfl_xor(l_run, 16, 32);
  float M = fmaxf(m_run, m_o);
  float L = l_run * fast_exp2((m_run - M) * L2E) +
            l_o   * fast_exp2((m_o   - M) * L2E);
  if (hi == 0) {
    mOut[(size_t)b * NN + j0 + col] = M;
    lOut[(size_t)b * NN + j0 + col] = L;
  }
}

// ---------------------------------------------------------------------------
// Pass 2: O = V * P, out = x + gamma * O / l.
// grid (N/128, B), block 512 (16 waves). P chunk exchanged through LDS as a
// packed B-operand image: BP[j][p] = pack(P[2p][j], P[2p+1][j]), stride 20
// dwords.  Producer: 1 ds_store_b128/wave.  Consumer: 2 ds_load_b128/wave.
// ---------------------------------------------------------------------------
__global__ __launch_bounds__(512) void attn_kernel(
    const unsigned short* __restrict__ qT, const unsigned short* __restrict__ kT,
    const unsigned short* __restrict__ vB,
    const float* __restrict__ mIn, const float* __restrict__ lIn,
    const float* __restrict__ x, const float* __restrict__ gamma,
    float* __restrict__ out) {
  __shared__ __align__(16) unsigned BP[128 * 20];

  int b    = blockIdx.y;
  int w    = threadIdx.x >> 5;
  int lane = threadIdx.x & 31;
  int col  = lane & 15;
  int hi   = lane >> 4;
  int jt   = w & 7;         // j sub-tile within block
  int half = w >> 3;        // n-half for S, c-half for O
  int j0l  = jt * 16;
  int j0   = blockIdx.x * 128 + j0l;
  int n_sub  = half * 16;
  int c_base = half * 128;

  // k tile for this wave's 16 columns: resident across the whole kernel.
  V16 Bk;
  {
    const unsigned short* kr = kT + ((size_t)b * NN + j0 + col) * DD + hi * 16;
    Bk.q[0] = ld128(kr);
    Bk.q[1] = ld128(kr + 8);
  }
  float m_j  = mIn[(size_t)b * NN + j0 + col];
  float invl = 1.0f / lIn[(size_t)b * NN + j0 + col];

  v8f acc[8];
#pragma unroll
  for (int ct = 0; ct < 8; ++ct)
    acc[ct] = (v8f){0.f, 0.f, 0.f, 0.f, 0.f, 0.f, 0.f, 0.f};

  for (int n0 = 0; n0 < NN; n0 += 32) {
    // ---- S tile (16 n-rows x 16 j-cols); write packed P pairs to LDS ----
    V16 Aq;
    {
      const unsigned short* qr =
          qT + ((size_t)b * NN + n0 + n_sub + col) * DD + hi * 16;
      Aq.q[0] = ld128(qr);
      Aq.q[1] = ld128(qr + 8);
    }
    v8f S = {0.f, 0.f, 0.f, 0.f, 0.f, 0.f, 0.f, 0.f};
    S = __builtin_amdgcn_wmma_f32_16x16x32_bf16(false, Aq.v, false, Bk.v,
                                                (short)0, S, false, false);
    B128 pk;
#pragma unroll
    for (int s = 0; s < 4; ++s) {
      float p0 = fast_exp2((S[2 * s]     - m_j) * L2E);
      float p1 = fast_exp2((S[2 * s + 1] - m_j) * L2E);
      pk.u[s] = pack_bf16(p0, p1);   // rows n_sub+hi*8+2s, +2s+1 of column j
    }
    *(B128*)&BP[(j0l + col) * 20 + (n_sub >> 1) + hi * 4] = pk;
    __syncthreads();

    // ---- B operand: 2x ds_load_b128 straight into operand registers ----
    V16 BPv;
    {
      const B128* pp = (const B128*)&BP[(j0l + col) * 20 + hi * 8];
      BPv.q[0] = pp[0];
      BPv.q[1] = pp[1];
    }

    // ---- accumulate O tiles: 8 channel tiles of 16 ----
#pragma unroll
    for (int ct = 0; ct < 8; ++ct) {
      int c0 = c_base + ct * 16;
      V16 Av;
      const unsigned short* vr =
          vB + ((size_t)b * CC + c0 + col) * (size_t)NN + n0 + hi * 16;
      Av.q[0] = ld128(vr);
      Av.q[1] = ld128(vr + 8);
      acc[ct] = __builtin_amdgcn_wmma_f32_16x16x32_bf16(false, Av.v, false, BPv.v,
                                                        (short)0, acc[ct],
                                                        false, false);
    }
    __syncthreads();
  }

  // ---- finalize: out = x + gamma * O / l ----
  float g = gamma[0];
#pragma unroll
  for (int ct = 0; ct < 8; ++ct) {
#pragma unroll
    for (int r = 0; r < 8; ++r) {
      int c = c_base + ct * 16 + hi * 8 + r;
      size_t idx = ((size_t)b * CC + c) * NN + j0 + col;
      out[idx] = x[idx] + g * (acc[ct][r] * invl);
    }
  }
}

// ---------------------------------------------------------------------------
extern "C" void kernel_launch(void* const* d_in, const int* in_sizes, int n_in,
                              void* d_out, int out_size, void* d_ws, size_t ws_size,
                              hipStream_t stream) {
  const float* x     = (const float*)d_in[0];
  const float* wq    = (const float*)d_in[1];
  const float* bq    = (const float*)d_in[2];
  const float* wk    = (const float*)d_in[3];
  const float* bk    = (const float*)d_in[4];
  const float* wv    = (const float*)d_in[5];
  const float* bv    = (const float*)d_in[6];
  const float* gamma = (const float*)d_in[7];
  float* out = (float*)d_out;

  char* ws = (char*)d_ws;
  size_t qT_off = 0;
  size_t kT_off = qT_off + (size_t)NB * NN * DD * 2;     // 1 MB
  size_t v_off  = kT_off + (size_t)NB * NN * DD * 2;     // 1 MB
  size_t m_off  = v_off  + (size_t)NB * CC * NN * 2;     // 8 MB
  size_t l_off  = m_off  + (size_t)NB * NN * 4;
  size_t w_off  = l_off  + (size_t)NB * NN * 4;
  unsigned short* qT  = (unsigned short*)(ws + qT_off);
  unsigned short* kT  = (unsigned short*)(ws + kT_off);
  unsigned short* vBw = (unsigned short*)(ws + v_off);
  float* mS = (float*)(ws + m_off);
  float* lS = (float*)(ws + l_off);
  unsigned short* wbf = (unsigned short*)(ws + w_off);   // 160 KB

  wcvt_kernel<<<dim3(WROWS), 256, 0, stream>>>(wq, wk, wv, wbf);
  proj_kernel<<<dim3(NN / 128, NB), 512, 0, stream>>>(
      x, bq, bk, bv, wbf, qT, kT, vBw);
  stats_kernel<<<dim3(NN / 128, NB), 256, 0, stream>>>(qT, kT, mS, lS);
  attn_kernel<<<dim3(NN / 128, NB), 512, 0, stream>>>(
      qT, kT, vBw, mS, lS, x, gamma, out);
}
